// ModelMPNN_75797582840080
// MI455X (gfx1250) — matrix operand order repaired and checked by reading, not compiled, not run on hardware
//
#include <hip/hip_runtime.h>
#include <hip/hip_bf16.h>
#include <math.h>

#define HIDN 64
#define INC 8
#define NNODES 20000
#define WIN 8
#define NBIG (NNODES*WIN)
#define NEDGE 1600000
#define EDEC 500000
#define DHC 143
#define KDEC 286
#define KDECP 288      // KDEC padded to 32
#define NDECP 144      // DHC padded to 16 (9 tiles)
#define EPSF 1e-5f

typedef __attribute__((ext_vector_type(16))) __bf16 v16bf;
typedef __attribute__((ext_vector_type(8)))  float  v8f;

union frag16 { v16bf v; uint4 u[2]; };

__device__ __forceinline__ float sigf(float x){ return 1.0f/(1.0f+__expf(-x)); }

// ---------------- dtype conversion / layout builders ----------------
__global__ void k_cvt_bf16(const float* __restrict__ s, __bf16* __restrict__ d, int n){
  int i = blockIdx.x*blockDim.x + threadIdx.x;
  if (i < n) d[i] = (__bf16)s[i];
}
// src [K,N] row-major -> dst Bt [Np,Kp] row-major (transpose + zero pad)
__global__ void k_build_bt(const float* __restrict__ s, __bf16* __restrict__ d,
                           int K, int N, int Kp, int Np){
  int i = blockIdx.x*blockDim.x + threadIdx.x;
  if (i < Np*Kp){
    int n = i / Kp, k = i % Kp;
    d[i] = (k < K && n < N) ? (__bf16)s[(size_t)k*N + n] : (__bf16)0.0f;
  }
}
// x [NBIG, 8] f32 -> xbf [NBIG, 32] bf16 zero-padded
__global__ void k_pad_x(const float* __restrict__ x, __bf16* __restrict__ d){
  int i = blockIdx.x*blockDim.x + threadIdx.x;
  if (i < NBIG*32){
    int r = i >> 5, c = i & 31;
    d[i] = (c < INC) ? (__bf16)x[(size_t)r*INC + c] : (__bf16)0.0f;
  }
}
// gather-concat decoder input: EE [EDEC, 288] bf16 (zero-padded past 286)
__global__ void k_build_ee(const __bf16* __restrict__ emb, const int* __restrict__ src,
                           const int* __restrict__ trg, __bf16* __restrict__ ee){
  long i = (long)blockIdx.x*blockDim.x + threadIdx.x;
  if (i < (long)EDEC*KDECP){
    int e = (int)(i / KDECP), k = (int)(i % KDECP);
    __bf16 v = (__bf16)0.0f;
    if (k < DHC)       v = emb[(size_t)src[e]*DHC + k];
    else if (k < KDEC) v = emb[(size_t)trg[e]*DHC + (k - DHC)];
    ee[i] = v;
  }
}

// ---------------- degree / normalization ----------------
__global__ void k_deg_init(float* deg, int n){
  int i = blockIdx.x*blockDim.x + threadIdx.x;
  if (i < n) deg[i] = 1.0f;           // self-loop weight
}
__global__ void k_deg_accum(const int* __restrict__ col, const float* __restrict__ ew,
                            float* deg, int ne){
  int e = blockIdx.x*blockDim.x + threadIdx.x;
  if (e < ne) atomicAdd(&deg[col[e]], ew[e]);
}
__global__ void k_deg_to_dis(float* deg, int n){
  int i = blockIdx.x*blockDim.x + threadIdx.x;
  if (i < n){
    float dv = deg[i];
    deg[i] = (dv > 0.0f) ? rsqrtf(fmaxf(dv, EPSF)) : 0.0f;
  }
}

// ------- bf16 WMMA matmul, 4-wide N register blocking: C[M,N] = A[M,K] @ Bt[N,K]^T -------
// Preconditions: M%16==0, N%64==0, K%32==0, 16B-aligned pointers -> branch-free b128 loads.
// min-waves-per-EU = 1: let the allocator keep all 4 B fragments in distinct registers.
__global__ __launch_bounds__(128, 1)
void k_mm_wmma(const __bf16* __restrict__ A, int lda,
               const __bf16* __restrict__ Bt, int ldk,
               float* __restrict__ C, int ldc,
               int M, int N, int K)
{
  int wid  = (blockIdx.x*blockDim.x + threadIdx.x) >> 5;
  int lane = threadIdx.x & 31;
  int tilesN4 = N >> 6;                      // 64-wide N blocks
  int tm = wid / tilesN4, tn4 = wid % tilesN4;
  if (tm * 16 >= M) return;
  int hi = (lane >> 4) & 1;
  const __bf16* Arow = A  + (size_t)(tm*16 + (lane & 15))*lda + (hi << 3);
  const __bf16* Brow = Bt + (size_t)(tn4*64 + (lane & 15))*ldk + (hi << 3);
  v8f acc[4] = {};
  for (int k0 = 0; k0 < K; k0 += 32){
    frag16 a, b[4];
    a.u[0] = *(const uint4*)(Arow + k0);
    a.u[1] = *(const uint4*)(Arow + k0 + 16);
    #pragma unroll
    for (int j = 0; j < 4; ++j){
      const __bf16* bp = Brow + (size_t)(j*16)*ldk + k0;
      b[j].u[0] = *(const uint4*)(bp);
      b[j].u[1] = *(const uint4*)(bp + 16);
    }
    #pragma unroll
    for (int j = 0; j < 4; ++j)
      acc[j] = __builtin_amdgcn_wmma_f32_16x16x32_bf16(false, a.v, false, b[j].v,
                                                       (short)0, acc[j], false, false);
  }
  #pragma unroll
  for (int j = 0; j < 4; ++j){
    int coln = tn4*64 + j*16 + (lane & 15);
    #pragma unroll
    for (int v = 0; v < 8; ++v){
      int m = tm*16 + v + (hi << 3);
      C[(size_t)m*ldc + coln] = acc[j][v];
    }
  }
}

// ---------------- GCN aggregation ----------------
__global__ void k_agg_init(const float* __restrict__ lin, const float* __restrict__ dis,
                           const float* __restrict__ bias, float* __restrict__ agg, int total){
  int i = blockIdx.x*blockDim.x + threadIdx.x;
  if (i < total){
    int n = i >> 6, ch = i & 63;
    float dv = dis[n];
    agg[i] = dv*dv*lin[i] + bias[ch];                   // self-loop term + bias
  }
}
__global__ void k_agg_edges(const int* __restrict__ rowi, const int* __restrict__ coli,
                            const float* __restrict__ ew, const float* __restrict__ dis,
                            const float* __restrict__ lin, float* __restrict__ agg, int ne){
  int idx = blockIdx.x*blockDim.x + threadIdx.x;
  if (idx < ne*16){
    int e = idx >> 4, chunk = idx & 15;
    int r = rowi[e], c = coli[e];
    float nm = dis[r]*ew[e]*dis[c];
    float4 hv = *(const float4*)(lin + (size_t)r*HIDN + chunk*4);
    float* ac = agg + (size_t)c*HIDN + chunk*4;
    atomicAdd(&ac[0], nm*hv.x);
    atomicAdd(&ac[1], nm*hv.y);
    atomicAdd(&ac[2], nm*hv.z);
    atomicAdd(&ac[3], nm*hv.w);
  }
}

// ---------------- BatchNorm (training stats over relu'd activations) ----------------
__global__ void k_bn_stats(const float* __restrict__ agg, float* __restrict__ sum,
                           float* __restrict__ sum2, int rows){
  int tid = blockIdx.x*blockDim.x + threadIdx.x;      // 64 * 1024 threads
  int ch = tid & 63, slice = tid >> 6;
  float s = 0.0f, s2 = 0.0f;
  for (int r = slice; r < rows; r += 1024){
    float v = fmaxf(agg[(size_t)r*HIDN + ch], 0.0f);
    s += v; s2 += v*v;
  }
  atomicAdd(&sum[ch], s);
  atomicAdd(&sum2[ch], s2);
}
__global__ void k_bn_apply(const float* __restrict__ agg, const float* __restrict__ sum,
                           const float* __restrict__ sum2, const float* __restrict__ g,
                           const float* __restrict__ be, __bf16* __restrict__ xc,
                           int coloff, int rows){
  int i = blockIdx.x*blockDim.x + threadIdx.x;
  if (i < rows*HIDN){
    int n = i >> 6, ch = i & 63;
    float inv = 1.0f / (float)rows;
    float m = sum[ch]*inv;
    float v = sum2[ch]*inv - m*m;
    float y = (fmaxf(agg[i], 0.0f) - m) * rsqrtf(v + EPSF) * g[ch] + be[ch];
    xc[(size_t)n*(2*HIDN) + coloff + ch] = (__bf16)y;
  }
}

// ---------------- LSTM pointwise step ----------------
__global__ void k_lstm_step(const float* __restrict__ Gt, const float* __restrict__ hW,
                            const float* __restrict__ bih, const float* __restrict__ bhh,
                            float* __restrict__ h, float* __restrict__ c,
                            __bf16* __restrict__ hbf, __bf16* __restrict__ ybf){
  int idx = blockIdx.x*blockDim.x + threadIdx.x;
  if (idx < NNODES*HIDN){
    int n = idx >> 6, j = idx & 63;
    size_t base = (size_t)n*(4*HIDN);
    float gi = Gt[base +           j] + hW[base +           j] + bih[          j] + bhh[          j];
    float gf = Gt[base +   HIDN + j] + hW[base +   HIDN + j] + bih[  HIDN + j] + bhh[  HIDN + j];
    float gg = Gt[base + 2*HIDN + j] + hW[base + 2*HIDN + j] + bih[2*HIDN + j] + bhh[2*HIDN + j];
    float go = Gt[base + 3*HIDN + j] + hW[base + 3*HIDN + j] + bih[3*HIDN + j] + bhh[3*HIDN + j];
    float cn = sigf(gf)*c[idx] + sigf(gi)*tanhf(gg);
    float hn = sigf(go)*tanhf(cn);
    c[idx] = cn; h[idx] = hn;
    hbf[idx] = (__bf16)hn;
    if (ybf) ybf[idx] = (__bf16)hn;
  }
}

// ---------------- node embedding packing ----------------
__global__ void k_pack_H(const float* __restrict__ h, __bf16* __restrict__ emb, int off){
  int i = blockIdx.x*blockDim.x + threadIdx.x;
  if (i < NNODES*HIDN){
    int n = i >> 6, ch = i & 63;
    emb[(size_t)n*DHC + off + ch] = (__bf16)h[i];
  }
}
__global__ void k_pack_S(const float* __restrict__ x, __bf16* __restrict__ emb){
  int n = blockIdx.x*blockDim.x + threadIdx.x;
  if (n < NNODES){
    #pragma unroll
    for (int j = 0; j < INC; ++j)
      emb[(size_t)n*DHC + 2*HIDN + j] = (__bf16)x[(size_t)n*INC + j];
    #pragma unroll
    for (int j = 1; j < WIN; ++j)
      emb[(size_t)n*DHC + 2*HIDN + INC - 1 + j] =
          (__bf16)x[((size_t)j*NNODES + n)*INC + (INC - 1)];
  }
}

// ------ fused edge decoder: 9 N-tiles per wave (single EE pass) -> relu -> LDS -> dot ------
__global__ __launch_bounds__(128, 1)
void k_decoder(const __bf16* __restrict__ ee, const __bf16* __restrict__ Wat,
               const float* __restrict__ dba, const float* __restrict__ dWb,
               const float* __restrict__ dbb, float* __restrict__ out, int ned)
{
  __shared__ float smem[4][16][148];
  int w = threadIdx.x >> 5, lane = threadIdx.x & 31;
  int wid = blockIdx.x*4 + w;
  int e0 = wid * 16;
  bool active = (e0 < ned);                  // wave-uniform
  int hi = (lane >> 4) & 1;
  const __bf16* Arow = ee  + (size_t)(e0 + (lane & 15))*KDECP + (hi << 3);
  const __bf16* Brow = Wat + (size_t)(lane & 15)*KDECP + (hi << 3);
  if (active){
    v8f acc[9] = {};
    for (int k0 = 0; k0 < KDECP; k0 += 32){
      frag16 a, b[9];
      a.u[0] = *(const uint4*)(Arow + k0);
      a.u[1] = *(const uint4*)(Arow + k0 + 16);
      #pragma unroll
      for (int nt = 0; nt < 9; ++nt){
        const __bf16* bp = Brow + (size_t)(nt*16)*KDECP + k0;
        b[nt].u[0] = *(const uint4*)(bp);
        b[nt].u[1] = *(const uint4*)(bp + 16);
      }
      #pragma unroll
      for (int nt = 0; nt < 9; ++nt)
        acc[nt] = __builtin_amdgcn_wmma_f32_16x16x32_bf16(false, a.v, false, b[nt].v,
                                                          (short)0, acc[nt], false, false);
    }
    #pragma unroll
    for (int nt = 0; nt < 9; ++nt){
      int n = nt*16 + (lane & 15);
      float bs = (n < DHC) ? dba[n] : 0.0f;
      #pragma unroll
      for (int v = 0; v < 8; ++v){
        int m = v + (hi << 3);
        smem[w][m][n] = fmaxf(acc[nt][v] + bs, 0.0f);   // n <= 143 < 148: in-bounds
      }
    }
  }
  __syncthreads();
  if (active && lane < 16){
    float acc2 = dbb[0];
    for (int q = 0; q < DHC; ++q) acc2 += smem[w][lane][q]*dWb[q];
    out[e0 + lane] = acc2;
  }
}

// ======================= host side =======================
static inline int cdiv(long a, long b){ return (int)((a + b - 1)/b); }

extern "C" void kernel_launch(void* const* d_in, const int* in_sizes, int n_in,
                              void* d_out, int out_size, void* d_ws, size_t ws_size,
                              hipStream_t stream)
{
  const float* x    = (const float*)d_in[0];
  const float* ea   = (const float*)d_in[1];
  const int*   ei   = (const int*)  d_in[2];    // [2, NEDGE]
  const int*   ewi  = (const int*)  d_in[3];    // [2, EDEC]
  const float* W1   = (const float*)d_in[4];
  const float* b1   = (const float*)d_in[5];
  const float* g1   = (const float*)d_in[6];
  const float* be1  = (const float*)d_in[7];
  const float* W2   = (const float*)d_in[8];
  const float* b2   = (const float*)d_in[9];
  const float* g2   = (const float*)d_in[10];
  const float* be2  = (const float*)d_in[11];
  const float* Wih1 = (const float*)d_in[12];   // [256,128] == [N,K]
  const float* Whh1 = (const float*)d_in[13];   // [256,64]  == [N,K]
  const float* bih1 = (const float*)d_in[14];
  const float* bhh1 = (const float*)d_in[15];
  const float* Wih2 = (const float*)d_in[16];   // [256,64]  == [N,K]
  const float* Whh2 = (const float*)d_in[17];   // [256,64]  == [N,K]
  const float* bih2 = (const float*)d_in[18];
  const float* bhh2 = (const float*)d_in[19];
  const float* dWa  = (const float*)d_in[20];   // [286,143] == [K,N]
  const float* dba  = (const float*)d_in[21];
  const float* dWb  = (const float*)d_in[22];   // [143]
  const float* dbb  = (const float*)d_in[23];
  float* out = (float*)d_out;

  // ---- workspace allocator ----
  char* wp = (char*)d_ws;
  auto alloc = [&](size_t bytes)->void*{
    void* p = (void*)wp; wp += (bytes + 255) & ~(size_t)255; return p;
  };
  float*  dis   = (float*) alloc((size_t)NBIG*4);
  float*  lin   = (float*) alloc((size_t)NBIG*HIDN*4);
  float*  agg   = (float*) alloc((size_t)NBIG*HIDN*4);
  __bf16* xc    = (__bf16*)alloc((size_t)NBIG*2*HIDN*2);
  float*  G     = (float*) alloc((size_t)NBIG*4*HIDN*4);
  float*  hW    = (float*) alloc((size_t)NNODES*4*HIDN*4);
  float*  h     = (float*) alloc((size_t)NNODES*HIDN*4);
  float*  c     = (float*) alloc((size_t)NNODES*HIDN*4);
  __bf16* hbf   = (__bf16*)alloc((size_t)NNODES*HIDN*2);
  __bf16* y1bf  = (__bf16*)alloc((size_t)NBIG*HIDN*2);
  __bf16* emb   = (__bf16*)alloc((size_t)NNODES*DHC*2);
  float*  bnS   = (float*) alloc(64*4);
  float*  bnS2  = (float*) alloc(64*4);
  __bf16* xbf   = (__bf16*)alloc((size_t)NBIG*32*2);          // K padded 8->32
  __bf16* W1b   = (__bf16*)alloc((size_t)64*32*2);            // Bt [64,32]
  __bf16* W2b   = (__bf16*)alloc((size_t)64*64*2);            // Bt [64,64]
  __bf16* Wih1b = (__bf16*)alloc((size_t)256*128*2);          // Bt [256,128]
  __bf16* Whh1b = (__bf16*)alloc((size_t)256*64*2);           // Bt [256,64]
  __bf16* Wih2b = (__bf16*)alloc((size_t)256*64*2);
  __bf16* Whh2b = (__bf16*)alloc((size_t)256*64*2);
  __bf16* Wat   = (__bf16*)alloc((size_t)NDECP*KDECP*2);      // Bt [144,288]
  __bf16* EE    = (__bf16*)alloc((size_t)EDEC*KDECP*2);       // gathered decoder input

  const int T = 256;
  auto mm = [&](const __bf16* A, int lda, const __bf16* Bt, int ldk,
                float* C, int ldc, int M, int N, int K){
    long waves = (long)(M/16) * (N/64);
    k_mm_wmma<<<cdiv(waves*32, 128), 128, 0, stream>>>(A, lda, Bt, ldk, C, ldc, M, N, K);
  };

  // ---- weight/input layout builders ----
  k_build_bt<<<cdiv(64*32,  T), T, 0, stream>>>(W1,  W1b,  INC, HIDN, 32, 64);
  k_build_bt<<<cdiv(64*64,  T), T, 0, stream>>>(W2,  W2b,  HIDN, HIDN, 64, 64);
  k_cvt_bf16<<<cdiv(256*128,T), T, 0, stream>>>(Wih1, Wih1b, 256*128);
  k_cvt_bf16<<<cdiv(256*64, T), T, 0, stream>>>(Whh1, Whh1b, 256*64);
  k_cvt_bf16<<<cdiv(256*64, T), T, 0, stream>>>(Wih2, Wih2b, 256*64);
  k_cvt_bf16<<<cdiv(256*64, T), T, 0, stream>>>(Whh2, Whh2b, 256*64);
  k_build_bt<<<cdiv(NDECP*KDECP, T), T, 0, stream>>>(dWa, Wat, KDEC, DHC, KDECP, NDECP);
  k_pad_x   <<<cdiv((long)NBIG*32, T), T, 0, stream>>>(x, xbf);

  // ---- symmetric GCN normalization ----
  const int* rowi = ei;            // edge_index[0]
  const int* coli = ei + NEDGE;    // edge_index[1]
  k_deg_init  <<<cdiv(NBIG, T), T, 0, stream>>>(dis, NBIG);
  k_deg_accum <<<cdiv(NEDGE,T), T, 0, stream>>>(coli, ea, dis, NEDGE);
  k_deg_to_dis<<<cdiv(NBIG, T), T, 0, stream>>>(dis, NBIG);

  // ---- GCN layer 1 ----
  mm(xbf, 32, W1b, 32, lin, HIDN, NBIG, HIDN, 32);
  k_agg_init <<<cdiv((long)NBIG*HIDN, T), T, 0, stream>>>(lin, dis, b1, agg, NBIG*HIDN);
  k_agg_edges<<<cdiv((long)NEDGE*16,  T), T, 0, stream>>>(rowi, coli, ea, dis, lin, agg, NEDGE);
  hipMemsetAsync(bnS, 0, 64*4, stream);
  hipMemsetAsync(bnS2,0, 64*4, stream);
  k_bn_stats<<<cdiv(64*1024, T), T, 0, stream>>>(agg, bnS, bnS2, NBIG);
  k_bn_apply<<<cdiv((long)NBIG*HIDN, T), T, 0, stream>>>(agg, bnS, bnS2, g1, be1, xc, 0, NBIG);

  // ---- GCN layer 2 (reads h1 = xc cols 0..63 as bf16, lda=128) ----
  mm(xc, 2*HIDN, W2b, 64, lin, HIDN, NBIG, HIDN, HIDN);
  k_agg_init <<<cdiv((long)NBIG*HIDN, T), T, 0, stream>>>(lin, dis, b2, agg, NBIG*HIDN);
  k_agg_edges<<<cdiv((long)NEDGE*16,  T), T, 0, stream>>>(rowi, coli, ea, dis, lin, agg, NEDGE);
  hipMemsetAsync(bnS, 0, 64*4, stream);
  hipMemsetAsync(bnS2,0, 64*4, stream);
  k_bn_stats<<<cdiv(64*1024, T), T, 0, stream>>>(agg, bnS, bnS2, NBIG);
  k_bn_apply<<<cdiv((long)NBIG*HIDN, T), T, 0, stream>>>(agg, bnS, bnS2, g2, be2, xc, HIDN, NBIG);

  // ---- LSTM 1: hoist input matmul over all 8 timesteps ----
  mm(xc, 2*HIDN, Wih1b, 2*HIDN, G, 4*HIDN, NBIG, 4*HIDN, 2*HIDN);
  hipMemsetAsync(h,   0, (size_t)NNODES*HIDN*4, stream);
  hipMemsetAsync(c,   0, (size_t)NNODES*HIDN*4, stream);
  hipMemsetAsync(hbf, 0, (size_t)NNODES*HIDN*2, stream);
  for (int t = 0; t < WIN; ++t){
    mm(hbf, HIDN, Whh1b, HIDN, hW, 4*HIDN, NNODES, 4*HIDN, HIDN);
    k_lstm_step<<<cdiv((long)NNODES*HIDN, T), T, 0, stream>>>(
        G + (size_t)t*NNODES*4*HIDN, hW, bih1, bhh1, h, c, hbf,
        y1bf + (size_t)t*NNODES*HIDN);
  }
  k_pack_H<<<cdiv((long)NNODES*HIDN, T), T, 0, stream>>>(h, emb, 0);

  // ---- LSTM 2 over y1 ----
  mm(y1bf, HIDN, Wih2b, HIDN, G, 4*HIDN, NBIG, 4*HIDN, HIDN);
  hipMemsetAsync(h,   0, (size_t)NNODES*HIDN*4, stream);
  hipMemsetAsync(c,   0, (size_t)NNODES*HIDN*4, stream);
  hipMemsetAsync(hbf, 0, (size_t)NNODES*HIDN*2, stream);
  for (int t = 0; t < WIN; ++t){
    mm(hbf, HIDN, Whh2b, HIDN, hW, 4*HIDN, NNODES, 4*HIDN, HIDN);
    k_lstm_step<<<cdiv((long)NNODES*HIDN, T), T, 0, stream>>>(
        G + (size_t)t*NNODES*4*HIDN, hW, bih2, bhh2, h, c, hbf, (__bf16*)nullptr);
  }
  k_pack_H<<<cdiv((long)NNODES*HIDN, T), T, 0, stream>>>(h, emb, HIDN);
  k_pack_S<<<cdiv(NNODES, T), T, 0, stream>>>(x, emb);

  // ---- fused edge decoder ----
  const int* src = ewi;
  const int* trg = ewi + EDEC;
  k_build_ee<<<cdiv((long)EDEC*KDECP, T), T, 0, stream>>>(emb, src, trg, EE);
  int dwaves = EDEC/16;
  k_decoder<<<cdiv(dwaves, 4), 128, 0, stream>>>(EE, Wat, dba, dWb, dbb, out, EDEC);
}
